// MultiHeadAttentionV2_43619687859008
// MI455X (gfx1250) — matrix operand order repaired and checked by reading, not compiled, not run on hardware
//
#include <hip/hip_runtime.h>
#include <hip/hip_bf16.h>

typedef __attribute__((ext_vector_type(8)))  __bf16 v8bf;
typedef __attribute__((ext_vector_type(16))) __bf16 v16bf;
typedef __attribute__((ext_vector_type(8)))  float  v8f;
typedef unsigned int u32x4 __attribute__((ext_vector_type(4)));
typedef int          i32x4 __attribute__((ext_vector_type(4)));
typedef int          i32x8 __attribute__((ext_vector_type(8)));

#define LOG2E 1.44269504088896340736f
#ifndef USE_TDM
#define USE_TDM 1
#endif

constexpr int B_ = 4, S_ = 2048, H_ = 8, D_ = 64;
constexpr int HD_ = H_ * D_;          // 512
constexpr int NROW = B_ * S_;         // 8192
constexpr int KPAD = 72;              // K tile LDS row stride (elems): 64 + 8
constexpr int VPAD = 40;              // V tile LDS row stride (elems): 32 + 8

// ---------- helpers ----------

__device__ __forceinline__ unsigned short f2bf(float f) {
    unsigned u = __float_as_uint(f);
    unsigned r = (u + 0x7FFFu + ((u >> 16) & 1u)) >> 16;
    return (unsigned short)r;
}

// A-fragment (16x32 bf16): lane<16 row=l16, elems 0..7 -> K k0+0..7, 8..15 -> K k0+16..23
//                          lane>=16: K k0+8..15 and k0+24..31
__device__ __forceinline__ v16bf load_fragA(const unsigned short* base, int row0, int ld,
                                            int k0, int l16, int lh) {
    const unsigned short* p = base + (size_t)(row0 + l16) * ld + k0 + lh * 8;
    v8bf lo = *reinterpret_cast<const v8bf*>(p);
    v8bf hi = *reinterpret_cast<const v8bf*>(p + 16);
    return __builtin_shufflevector(lo, hi, 0,1,2,3,4,5,6,7,8,9,10,11,12,13,14,15);
}

// B-fragment (32x16 bf16), source stored transposed (rows = N, contiguous over K):
// lane<16 col=l16 holds K k0+0..7 and k0+16..23; lane>=16 holds K k0+8..15 and k0+24..31
__device__ __forceinline__ v16bf load_fragB(const unsigned short* base, int row0, int ld,
                                            int k0, int l16, int lh) {
    const unsigned short* p = base + (size_t)(row0 + l16) * ld + k0 + lh * 8;
    v8bf lo = *reinterpret_cast<const v8bf*>(p);
    v8bf hi = *reinterpret_cast<const v8bf*>(p + 16);
    return __builtin_shufflevector(lo, hi, 0,1,2,3,4,5,6,7,8,9,10,11,12,13,14,15);
}

__device__ __forceinline__ v8f wmma_bf16(v16bf a, v16bf b, v8f c) {
    return __builtin_amdgcn_wmma_f32_16x16x32_bf16(false, a, false, b, (short)0, c,
                                                   false, false);
}

#if USE_TDM
// Issue one 2D TDM tile load (bf16 elements) global -> LDS, with LDS-side padding.
// D# per CDNA5 ISA: group0 = {flags, lds_addr, global_addr lo/hi|type=2},
// group1 = dims/strides/pad. Groups 2/3 zero (<=2D tensor).
// Toolchain uses the 6-arg builtin: (u32x4, i32x8, i32x4, i32x4, i32x8, i32 cpol).
__device__ __forceinline__ void tdm_load_2d(unsigned lds_off, const void* gsrc,
                                            unsigned tensor_d0, unsigned tensor_d1,
                                            unsigned tile_d0, unsigned tile_d1,
                                            unsigned long long stride0,
                                            unsigned pad_interval_code,
                                            unsigned pad_amount_code) {
    unsigned long long ga = (unsigned long long)(size_t)gsrc;
    u32x4 g0 = { 1u,                                          // count=1 (valid user D#)
                 lds_off,                                     // lds_addr (bytes)
                 (unsigned)ga,                                // global_addr[31:0]
                 (unsigned)((ga >> 32) & 0x01FFFFFFu) | (2u << 30) };  // addr hi | type=2
    unsigned w0 = (1u << 16)                                  // data_size = 2 bytes
                | (1u << 20)                                  // pad_enable
                | (pad_interval_code << 22)
                | (pad_amount_code << 25);
    i32x8 g1 = { (int)w0,
                 (int)((tensor_d0 & 0xFFFFu) << 16),                          // dim0 lo16
                 (int)(((tensor_d0 >> 16) & 0xFFFFu) | ((tensor_d1 & 0xFFFFu) << 16)),
                 (int)(((tensor_d1 >> 16) & 0xFFFFu) | ((tile_d0 & 0xFFFFu) << 16)),
                 (int)(tile_d1 & 0xFFFFu),                                    // tile_dim1, tile_dim2=0
                 (int)(unsigned)(stride0 & 0xFFFFFFFFu),                      // dim0 stride lo
                 (int)(unsigned)((stride0 >> 32) & 0xFFFFu),                  // dim0 stride hi
                 0 };
    i32x4 z4 = { 0, 0, 0, 0 };
    i32x8 z8 = { 0, 0, 0, 0, 0, 0, 0, 0 };
    __builtin_amdgcn_tensor_load_to_lds(g0, g1, z4, z4, z8, 0);
}
#endif

// ---------- stage 0: conversions / transposes ----------

__global__ __launch_bounds__(256) void cvt_x_kernel(const float* __restrict__ x,
                                                    unsigned short* __restrict__ kbf, int n) {
    int i = blockIdx.x * 256 + threadIdx.x;
    if (i < n) kbf[i] = f2bf(x[i]);
}

// w [rows x cols] f32 -> wt [cols x rows] bf16
__global__ __launch_bounds__(256) void transpose_w_kernel(const float* __restrict__ w,
                                                          unsigned short* __restrict__ wt,
                                                          int rows, int cols) {
    int i = blockIdx.x * 256 + threadIdx.x;
    if (i < rows * cols) {
        int r = i / cols, c = i % cols;
        wt[(size_t)c * rows + r] = f2bf(w[i]);
    }
}

// ---------- stage 1: QV projection (x @ Wqv): q scaled, v transposed; branchless scatter ----------

__global__ __launch_bounds__(256) void qv_gemm_kernel(const unsigned short* __restrict__ xbf,
                                                      const unsigned short* __restrict__ wqvT,
                                                      unsigned short* __restrict__ qbf,
                                                      unsigned short* __restrict__ vTbf) {
    int lane = threadIdx.x & 31, wave = threadIdx.x >> 5;
    int l16 = lane & 15, lh = lane >> 4;
    int m0 = blockIdx.x * 128 + wave * 16;     // row tile in [0, 8192)
    int ng = blockIdx.y * 64;                  // col group in [0, 1024)

    v16bf a0 = load_fragA(xbf, m0, D_, 0,  l16, lh);
    v16bf a1 = load_fragA(xbf, m0, D_, 32, l16, lh);

    int b = m0 >> 11;                 // 16-row tile never crosses a batch boundary
    int sbase = (m0 & (S_ - 1)) + lh * 8;

    // load all B fragments, then issue independent WMMAs back-to-back (no D->C stalls)
    v16bf b0[4], b1[4];
#pragma unroll
    for (int nt = 0; nt < 4; ++nt) {
        b0[nt] = load_fragB(wqvT, ng + nt * 16, D_, 0,  l16, lh);
        b1[nt] = load_fragB(wqvT, ng + nt * 16, D_, 32, l16, lh);
    }
    v8f acc[4];
#pragma unroll
    for (int nt = 0; nt < 4; ++nt) { v8f z = {}; acc[nt] = wmma_bf16(a0, b0[nt], z); }
#pragma unroll
    for (int nt = 0; nt < 4; ++nt) acc[nt] = wmma_bf16(a1, b1[nt], acc[nt]);

#pragma unroll
    for (int nt = 0; nt < 4; ++nt) {
        int c = ng + nt * 16 + l16;             // global qv channel
        int t = c & 1, d = (c >> 1) & 63, head = c >> 7;
        bool isq = (t == 0);
        // branchless scatter: q -> [b,head,s,d] (stride D over s), v -> [b,head,d,s] (stride 1)
        unsigned short* dst = isq ? qbf : vTbf;
        size_t base = isq ? ((((size_t)(b * H_ + head)) * S_) * D_ + d)
                          : ((((size_t)(b * H_ + head)) * D_ + d) * S_);
        size_t stride = isq ? (size_t)D_ : (size_t)1;
        float fac = isq ? 0.125f : 1.0f;
#pragma unroll
        for (int i = 0; i < 8; ++i)
            dst[base + (size_t)(sbase + i) * stride] = f2bf(acc[nt][i] * fac);
    }
}

// ---------- stage 2: flash attention with LDS-staged (TDM) K/V tiles ----------

__global__ __launch_bounds__(256) void attn_kernel(const unsigned short* __restrict__ qbf,
                                                   const unsigned short* __restrict__ kbf,
                                                   const unsigned short* __restrict__ vTbf,
                                                   unsigned short* __restrict__ obf) {
    __shared__ unsigned short ldsK[2][32 * KPAD];   // 32 keys x 64 d (padded rows)
    __shared__ unsigned short ldsV[2][64 * VPAD];   // 64 d    x 32 keys (padded rows)
    __shared__ unsigned short pLds[8][16 * VPAD];   // per-wave P tile 16x32

    int tid = threadIdx.x;
    int lane = tid & 31, wave = tid >> 5;
    int l16 = lane & 15, lh = lane >> 4;
    int bh = blockIdx.y;                 // b*H + h
    int b = bh >> 3, h = bh & 7;
    int m0 = blockIdx.x * 128 + wave * 16;   // query row tile within S

    const unsigned short* qB = qbf  + ((size_t)bh * S_) * D_;
    const unsigned short* kB = kbf  + ((size_t)b  * S_) * D_;
    const unsigned short* vB = vTbf + ((size_t)bh * D_) * S_;
    unsigned short* pW = &pLds[wave][0];

    v16bf aq0 = load_fragA(qB, m0, D_, 0,  l16, lh);
    v16bf aq1 = load_fragA(qB, m0, D_, 32, l16, lh);

    float mrow[8], lrow[8];
    v8f o[4] = {};
#pragma unroll
    for (int i = 0; i < 8; ++i) { mrow[i] = -1e30f; lrow[i] = 0.0f; }

#if USE_TDM
    // prologue: stage tile 0 into buffer 0 (2 TDM ops; pad codes give 72/40 elem strides)
    if (wave == 0) {
        tdm_load_2d((unsigned)(size_t)&ldsK[0][0], kB, D_, S_, D_, 32, D_, 4, 3);
        tdm_load_2d((unsigned)(size_t)&ldsV[0][0], vB, S_, D_, 32, D_, S_, 3, 3);
    }
#else
    {   // cooperative staging, 16B vectors
        int kr = tid >> 3, kc = (tid & 7) * 8;
        *(v8bf*)(&ldsK[0][kr * KPAD + kc]) = *(const v8bf*)(kB + (size_t)kr * D_ + kc);
        int vr = tid >> 2, vc = (tid & 3) * 8;
        *(v8bf*)(&ldsV[0][vr * VPAD + vc]) = *(const v8bf*)(vB + (size_t)vr * S_ + vc);
    }
#endif

    for (int it = 0; it < S_ / 32; ++it) {
        int t0 = it * 32;
        int cur = it & 1, nxt = cur ^ 1;

#if USE_TDM
        if (wave == 0) {
            if (t0 + 32 < S_) {
                tdm_load_2d((unsigned)(size_t)&ldsK[nxt][0], kB + (size_t)(t0 + 32) * D_,
                            D_, S_, D_, 32, D_, 4, 3);
                tdm_load_2d((unsigned)(size_t)&ldsV[nxt][0], vB + (t0 + 32),
                            S_, D_, 32, D_, S_, 3, 3);
                __builtin_amdgcn_s_wait_tensorcnt(2);   // retire current tile's 2 ops
            } else {
                __builtin_amdgcn_s_wait_tensorcnt(0);
            }
        }
        __syncthreads();
#else
        if (t0 + 32 < S_) {
            int kr = tid >> 3, kc = (tid & 7) * 8;
            *(v8bf*)(&ldsK[nxt][kr * KPAD + kc]) =
                *(const v8bf*)(kB + (size_t)(t0 + 32 + kr) * D_ + kc);
            int vr = tid >> 2, vc = (tid & 3) * 8;
            *(v8bf*)(&ldsV[nxt][vr * VPAD + vc]) =
                *(const v8bf*)(vB + (size_t)vr * S_ + t0 + 32 + vc);
        }
        __syncthreads();
#endif

        const unsigned short* kT = &ldsK[cur][0];
        const unsigned short* vT = &ldsV[cur][0];

        // QK^T: interleave the two accumulator chains so adjacent WMMAs are independent
        v16bf bk00 = load_fragB(kT, 0,  KPAD, 0,  l16, lh);
        v16bf bk10 = load_fragB(kT, 16, KPAD, 0,  l16, lh);
        v16bf bk01 = load_fragB(kT, 0,  KPAD, 32, l16, lh);
        v16bf bk11 = load_fragB(kT, 16, KPAD, 32, l16, lh);
        v8f s0 = {}, s1 = {};
        s0 = wmma_bf16(aq0, bk00, s0);
        s1 = wmma_bf16(aq0, bk10, s1);
        s0 = wmma_bf16(aq1, bk01, s0);
        s1 = wmma_bf16(aq1, bk11, s1);

        float scale[8];
#pragma unroll
        for (int i = 0; i < 8; ++i) {
            float tm = fmaxf(s0[i], s1[i]);
#pragma unroll
            for (int mk = 1; mk < 16; mk <<= 1) tm = fmaxf(tm, __shfl_xor(tm, mk, 32));
            float mnew = fmaxf(mrow[i], tm);
            float sc = exp2f((mrow[i] - mnew) * LOG2E);
            float p0 = exp2f((s0[i] - mnew) * LOG2E);
            float p1 = exp2f((s1[i] - mnew) * LOG2E);
            float rs = p0 + p1;
#pragma unroll
            for (int mk = 1; mk < 16; mk <<= 1) rs += __shfl_xor(rs, mk, 32);
            lrow[i] = lrow[i] * sc + rs;
            mrow[i] = mnew;
            scale[i] = sc;
            unsigned short* pr = pW + (lh * 8 + i) * VPAD;
            pr[l16]      = f2bf(p0);
            pr[16 + l16] = f2bf(p1);
        }
#pragma unroll
        for (int i = 0; i < 8; ++i) {
            o[0][i] *= scale[i]; o[1][i] *= scale[i];
            o[2][i] *= scale[i]; o[3][i] *= scale[i];
        }

        asm volatile("s_wait_dscnt 0" ::: "memory");   // P tile C->A layout turnaround
        v16bf aP = load_fragA(pW, 0, VPAD, 0, l16, lh);

#pragma unroll
        for (int nt = 0; nt < 4; ++nt) {
            v16bf bv = load_fragB(vT, nt * 16, VPAD, 0, l16, lh);
            o[nt] = wmma_bf16(aP, bv, o[nt]);
        }
        __syncthreads();   // all waves done reading cur before it is overwritten
    }

#pragma unroll
    for (int i = 0; i < 8; ++i) {
        float inv = 1.0f / lrow[i];
        int s = m0 + lh * 8 + i;
        size_t rowbase = ((size_t)(b * S_ + s)) * HD_ + h * D_;
#pragma unroll
        for (int nt = 0; nt < 4; ++nt)
            obf[rowbase + nt * 16 + l16] = f2bf(o[nt][i] * inv);
    }
}

// ---------- stage 3: output projection + bias ----------

__global__ __launch_bounds__(256) void out_gemm_kernel(const unsigned short* __restrict__ obf,
                                                       const unsigned short* __restrict__ wcT,
                                                       const float* __restrict__ bc,
                                                       float* __restrict__ out) {
    int lane = threadIdx.x & 31, wave = threadIdx.x >> 5;
    int l16 = lane & 15, lh = lane >> 4;
    int m0 = blockIdx.x * 128 + wave * 16;

    v8f acc[4] = {};
    for (int k0 = 0; k0 < HD_; k0 += 32) {
        v16bf a = load_fragA(obf, m0, HD_, k0, l16, lh);
#pragma unroll
        for (int nt = 0; nt < 4; ++nt) {
            v16bf bb = load_fragB(wcT, nt * 16, HD_, k0, l16, lh);
            acc[nt] = wmma_bf16(a, bb, acc[nt]);
        }
    }
#pragma unroll
    for (int nt = 0; nt < 4; ++nt) {
        int col = nt * 16 + l16;
        float bias = bc[col];
#pragma unroll
        for (int i = 0; i < 8; ++i) {
            int m = m0 + lh * 8 + i;
            out[(size_t)m * D_ + col] = acc[nt][i] + bias;
        }
    }
}

// ---------- launch ----------

extern "C" void kernel_launch(void* const* d_in, const int* in_sizes, int n_in,
                              void* d_out, int out_size, void* d_ws, size_t ws_size,
                              hipStream_t stream) {
    (void)in_sizes; (void)n_in; (void)out_size; (void)ws_size;
    const float* x   = (const float*)d_in[0];
    const float* Wqv = (const float*)d_in[1];
    const float* Wc  = (const float*)d_in[2];
    const float* bc  = (const float*)d_in[3];
    float* out = (float*)d_out;

    char* ws = (char*)d_ws;
    unsigned short* kbf  = (unsigned short*)(ws);                               // 1 MB  [B*S, 64]
    unsigned short* qbf  = (unsigned short*)(ws + (size_t)1  * (1u << 20));     // 8 MB  [B,H,S,D]
    unsigned short* vTbf = (unsigned short*)(ws + (size_t)9  * (1u << 20));     // 8 MB  [B,H,D,S]
    unsigned short* obf  = (unsigned short*)(ws + (size_t)17 * (1u << 20));     // 8 MB  [B*S, 512]
    unsigned short* wqvT = (unsigned short*)(ws + (size_t)25 * (1u << 20));     // 128 KB [1024, 64]
    unsigned short* wcT  = (unsigned short*)(ws + (size_t)25 * (1u << 20) + (1u << 17)); // 64 KB

    int nX = NROW * D_;
    cvt_x_kernel<<<(nX + 255) / 256, 256, 0, stream>>>(x, kbf, nX);
    transpose_w_kernel<<<(D_ * 2 * HD_ + 255) / 256, 256, 0, stream>>>(Wqv, wqvT, D_, 2 * HD_);
    transpose_w_kernel<<<(HD_ * D_ + 255) / 256, 256, 0, stream>>>(Wc, wcT, HD_, D_);

    qv_gemm_kernel<<<dim3(NROW / 128, (2 * HD_) / 64), 256, 0, stream>>>(kbf, wqvT, qbf, vTbf);
    attn_kernel<<<dim3(S_ / 128, B_ * H_), 256, 0, stream>>>(qbf, kbf, vTbf, obf);
    out_gemm_kernel<<<NROW / 128, 256, 0, stream>>>(obf, wcT, bc, out);
}
